// OptimizedLanguageModel_32701880991973
// MI455X (gfx1250) — matrix operand order repaired and checked by reading, not compiled, not run on hardware
//
#include <hip/hip_runtime.h>
#include <hip/hip_bf16.h>

// ---------------------------------------------------------------------------
// Model dims (fixed by the reference)
// ---------------------------------------------------------------------------
constexpr int B_  = 2;
constexpr int T_  = 2048;
constexpr int DE_ = 1024;
constexpr int H_  = 2048;
constexpr int NH_ = 16;
constexpr int KD_ = 128;
constexpr int VD_ = 128;
constexpr int V_  = 32000;
constexpr int M_  = B_ * T_;                    // 4096 rows
constexpr int QKV_ = NH_ * (2 * KD_ + VD_);     // 6144

typedef __attribute__((ext_vector_type(16))) __bf16 v16bf;
typedef __attribute__((ext_vector_type(8)))  __bf16 v8bf;
typedef __attribute__((ext_vector_type(8)))  float  v8f;

// ---------------------------------------------------------------------------
// WMMA helpers (CDNA5 wave32, D = A*B + C, f32 accumulate)
// ---------------------------------------------------------------------------
__device__ __forceinline__ v8f wmma_bf16(v16bf a, v16bf b, v8f c) {
  return __builtin_amdgcn_wmma_f32_16x16x32_bf16(false, a, false, b, (short)0, c,
                                                 false, false);
}

__device__ __forceinline__ v16bf frag_cat(v8bf lo, v8bf hi) {
  v16bf r;
#pragma unroll
  for (int i = 0; i < 8; ++i) { r[i] = lo[i]; r[i + 8] = hi[i]; }
  return r;
}

// A operand (16xK tile, row = lane&15): elems 0..7 -> K = h*8..h*8+7,
// elems 8..15 -> K = 16+h*8.. (ISA 7.12.2, 16-bit A 16x32)
__device__ __forceinline__ v16bf load_frag_a(const __bf16* base, int ld, int lane) {
  const int row = lane & 15, half = lane >> 4;
  const __bf16* p = base + (size_t)row * ld + half * 8;
  return frag_cat(*(const v8bf*)p, *(const v8bf*)(p + 16));
}

// B operand (Kx16 tile stored as [n][K] rows): elems 0..7 -> K = h*8..h*8+7,
// elems 8..15 -> K = 16+h*8.. (same order as A)
__device__ __forceinline__ v16bf load_frag_b(const __bf16* base, int ld, int lane) {
  const int col = lane & 15, half = lane >> 4;
  const __bf16* p = base + (size_t)col * ld + half * 8;
  return frag_cat(*(const v8bf*)p, *(const v8bf*)(p + 16));
}

__device__ __forceinline__ float gelu_f(float x) {
  return 0.5f * x * (1.0f + erff(x * 0.70710678118654752f));
}

// ---------------------------------------------------------------------------
// CDNA5 async global -> LDS copy (GLOBAL_LOAD_ASYNC_TO_LDS_B128, ASYNCcnt).
// Per-lane: 16 bytes from lane's global address to lane's LDS byte address.
// ---------------------------------------------------------------------------
#define AS3 __attribute__((address_space(3)))
__device__ __forceinline__ unsigned lds_off_u32(void* p) {
  return (unsigned)(uintptr_t)(AS3 void*)p;
}

__device__ __forceinline__ void async_copy_b128(unsigned lds_byte_off,
                                                const void* gptr) {
  asm volatile("global_load_async_to_lds_b128 %0, %1, off"
               :
               : "v"(lds_byte_off), "v"((unsigned long long)(uintptr_t)gptr)
               : "memory");
}

// ---------------------------------------------------------------------------
// Small utility kernels
// ---------------------------------------------------------------------------
__global__ void cvt_f32_bf16_kernel(const float* __restrict__ s,
                                    __bf16* __restrict__ d, size_t n) {
  size_t i = (size_t)blockIdx.x * blockDim.x + threadIdx.x;
  size_t stride = (size_t)gridDim.x * blockDim.x;
  for (; i < n; i += stride) d[i] = (__bf16)s[i];
}

__global__ void zero_u32_kernel(unsigned int* __restrict__ p, int n) {
  int i = blockIdx.x * blockDim.x + threadIdx.x;
  if (i < n) p[i] = 0u;
}

__global__ void embed_gather_kernel(const int* __restrict__ tokens,
                                    const float* __restrict__ E,
                                    __bf16* __restrict__ xb) {
  const int row = blockIdx.x;
  const int tok = tokens[row];
  const size_t src = (size_t)tok * DE_;
  const size_t dst = (size_t)row * DE_;
  for (int c = threadIdx.x; c < DE_; c += blockDim.x)
    xb[dst + c] = (__bf16)E[src + c];
}

// ---------------------------------------------------------------------------
// bf16 WMMA GEMM:  C[M,N] = A[M,K] * W[N,K]^T (+bias)
// 128x128 block tile, BK=32, 256 threads = 8 waves (4x2 of 32x64).
// Double-buffered LDS filled by async global->LDS DMA; compute of stage i
// overlaps the 4 outstanding async copies of stage i+1 (s_wait_asynccnt 4).
// M, N multiples of 128 and K multiple of 32 -> no predication (EXEC full).
// ---------------------------------------------------------------------------
template <bool HAS_BIAS>
__global__ void gemm_bf16_kernel(const __bf16* __restrict__ A,
                                 const __bf16* __restrict__ W,
                                 const float* __restrict__ bias,
                                 float* __restrict__ Cf,
                                 __bf16* __restrict__ Cb,
                                 int M, int N, int K) {
  constexpr int LDT = 40;   // 32 + 8 bf16 pad (80B rows, 16B aligned)
  __shared__ __align__(16) __bf16 As[2][128 * LDT];
  __shared__ __align__(16) __bf16 Ws[2][128 * LDT];

  const int bm = blockIdx.y * 128;
  const int bn = blockIdx.x * 128;
  const int tid = threadIdx.x;
  const int wave = tid >> 5, lane = tid & 31;
  const int half = lane >> 4, ln = lane & 15;
  const int wm = (wave >> 1) * 32;     // 4 row groups of 32
  const int wn = (wave & 1) * 64;      // 2 col groups of 64

  v8f acc[2][4] = {};

  const int r  = tid >> 1;             // 0..127: tile row
  const int cs = (tid & 1) * 16;       // 0 or 16: K segment

  // issue one stage (4 async b128 per thread: 2 for A tile, 2 for W tile)
  auto issue_stage = [&](int k0, int buf) {
    const __bf16* ga = A + (size_t)(bm + r) * K + k0 + cs;
    const __bf16* gw = W + (size_t)(bn + r) * K + k0 + cs;
    const unsigned la = lds_off_u32(&As[buf][r * LDT + cs]);
    const unsigned lw = lds_off_u32(&Ws[buf][r * LDT + cs]);
    __builtin_prefetch(ga + 64, 0, 3);     // pre-warm L2 two stages ahead
    __builtin_prefetch(gw + 64, 0, 3);
    async_copy_b128(la,      ga);
    async_copy_b128(la + 16, ga + 8);
    async_copy_b128(lw,      gw);
    async_copy_b128(lw + 16, gw + 8);
  };

  issue_stage(0, 0);
  int buf = 0;
  for (int k0 = 0; k0 < K; k0 += 32) {
    const bool more = (k0 + 32) < K;
    if (more) {
      issue_stage(k0 + 32, buf ^ 1);
      asm volatile("s_wait_asynccnt 0x4" ::: "memory");  // current stage done
    } else {
      asm volatile("s_wait_asynccnt 0x0" ::: "memory");
    }
    __syncthreads();

    v16bf af[2], wf[4];
#pragma unroll
    for (int i = 0; i < 2; ++i)
      af[i] = load_frag_a(&As[buf][(wm + i * 16) * LDT], LDT, lane);
#pragma unroll
    for (int j = 0; j < 4; ++j)
      wf[j] = load_frag_b(&Ws[buf][(wn + j * 16) * LDT], LDT, lane);
#pragma unroll
    for (int i = 0; i < 2; ++i)
#pragma unroll
      for (int j = 0; j < 4; ++j)
        acc[i][j] = wmma_bf16(af[i], wf[j], acc[i][j]);

    __syncthreads();
    buf ^= 1;
  }

#pragma unroll
  for (int i = 0; i < 2; ++i) {
#pragma unroll
    for (int j = 0; j < 4; ++j) {
      const int coln = bn + wn + j * 16 + ln;
      const float bv = HAS_BIAS ? bias[coln] : 0.0f;
#pragma unroll
      for (int rr = 0; rr < 8; ++rr) {
        const int rowm = bm + wm + i * 16 + rr + 8 * half;
        const float v = acc[i][j][rr] + bv;
        if (Cf) Cf[(size_t)rowm * N + coln] = v;
        if (Cb) Cb[(size_t)rowm * N + coln] = (__bf16)v;
      }
    }
  }
}

// ---------------------------------------------------------------------------
// RNN scan: h_t = gelu(xi_t + h_{t-1} @ Wh^T), sequential over T.
// Persistent cooperative kernel: 128 single-wave WGs x 16 output columns,
// batch (B=2) padded to M=16 rows so the per-step GEMM is WMMA.
// Grid sync via monotonically increasing atomic counter.
// ---------------------------------------------------------------------------
__global__ void rnn_scan_kernel(const float* __restrict__ xi,
                                const __bf16* __restrict__ Whb,
                                __bf16* __restrict__ hbuf,     // 2 * 16*H
                                __bf16* __restrict__ hs,       // (B*T, H) bf16
                                unsigned int* __restrict__ bar) {
  const int lane = threadIdx.x & 31;
  const int half = lane >> 4, ln = lane & 15;
  const int n0 = blockIdx.x * 16;
  const int col = n0 + ln;
  const unsigned nwg = gridDim.x;

  __bf16* buf0 = hbuf;
  __bf16* buf1 = hbuf + 16 * H_;

  for (int t = 0; t < T_; ++t) {
    const __bf16* hcur = (t & 1) ? buf1 : buf0;
    __bf16*       hnxt = (t & 1) ? buf0 : buf1;

    v8f acc = {};
    for (int kc = 0; kc < H_; kc += 32) {
      v16bf a = load_frag_a(hcur + kc, H_, lane);
      v16bf w = load_frag_b(Whb + (size_t)n0 * H_ + kc, H_, lane);
      acc = wmma_bf16(a, w, acc);
    }

#pragma unroll
    for (int rr = 0; rr < 8; ++rr) {
      const int m = rr + 8 * half;                 // padded batch row
      const float xv = (m < B_) ? xi[((size_t)m * T_ + t) * H_ + col] : 0.0f;
      const float h2 = gelu_f(acc[rr] + xv);
      hnxt[m * H_ + col] = (__bf16)h2;             // rows >= B stay exactly 0
      if (m < B_) hs[((size_t)m * T_ + t) * H_ + col] = (__bf16)h2;
    }

    // device-wide step barrier
    __threadfence();
    if (threadIdx.x == 0) {
      __hip_atomic_fetch_add(bar, 1u, __ATOMIC_RELEASE, __HIP_MEMORY_SCOPE_AGENT);
      const unsigned target = nwg * (unsigned)(t + 1);
      while (__hip_atomic_load(bar, __ATOMIC_ACQUIRE, __HIP_MEMORY_SCOPE_AGENT) < target)
        __builtin_amdgcn_s_sleep(2);
    }
    __threadfence();
    __syncthreads();
  }
}

// ---------------------------------------------------------------------------
// Causal flash attention. Block = 4 waves, one (b,h) head, 64 q rows
// (16 per wave). K/V tiles (32 keys) staged in LDS, V transposed for the
// P*V B-operand. Online softmax per wave; P spills through per-wave LDS
// with an explicit s_wait_dscnt before reload.
// ---------------------------------------------------------------------------
__global__ void attn_kernel(const __bf16* __restrict__ qkv,
                            __bf16* __restrict__ ao) {
  __shared__ __align__(16) __bf16 kt[32 * 128];   // [key][d]
  __shared__ __align__(16) __bf16 vT[128 * 32];   // [d][key]
  __shared__ __align__(16) __bf16 ps[4][16 * 32]; // per-wave P tile

  const int tid = threadIdx.x;
  const int wave = tid >> 5, lane = tid & 31;
  const int half = lane >> 4, ln = lane & 15;
  const int bh = blockIdx.x;
  const int b = bh / NH_, h = bh % NH_;
  const int q0b = blockIdx.y * 64;
  const int q0 = q0b + wave * 16;
  const size_t rowbase = (size_t)b * T_;
  const int qoff = h * KD_;
  const int koff = NH_ * KD_ + h * KD_;
  const int voff = 2 * NH_ * KD_ + h * VD_;
  const float scale = 0.088388347648318447f;       // 1/sqrt(128)

  v16bf qf[4];
#pragma unroll
  for (int c = 0; c < 4; ++c)
    qf[c] = load_frag_a(qkv + (rowbase + q0) * QKV_ + qoff + c * 32, QKV_, lane);

  v8f acc[8] = {};
  float mrow[8], lrow[8];
#pragma unroll
  for (int rr = 0; rr < 8; ++rr) { mrow[rr] = -INFINITY; lrow[rr] = 0.0f; }

  const int nkb = q0b / 32 + 2;                    // causal: keys <= q0b+63
  for (int kb = 0; kb < nkb; ++kb) {
    const int k0 = kb * 32;
    {                                              // cooperative K/V tile load
      const int r = tid >> 2, cseg = (tid & 3) * 32;
      const __bf16* kg = qkv + (rowbase + k0 + r) * (size_t)QKV_ + koff + cseg;
      v8bf* dk = (v8bf*)&kt[r * 128 + cseg];
#pragma unroll
      for (int i = 0; i < 4; ++i) dk[i] = *(const v8bf*)(kg + i * 8);
      const __bf16* vg = qkv + (rowbase + k0 + r) * (size_t)QKV_ + voff + cseg;
#pragma unroll
      for (int i = 0; i < 32; ++i) vT[(cseg + i) * 32 + r] = vg[i];
    }
    __syncthreads();

    if (k0 <= q0 + 15) {                           // wave-uniform -> EXEC full
      v8f s0 = {}, s1 = {};
#pragma unroll
      for (int c = 0; c < 4; ++c) {
        v16bf kf0 = load_frag_b(kt + 0 * 128 + c * 32, 128, lane);
        v16bf kf1 = load_frag_b(kt + 16 * 128 + c * 32, 128, lane);
        s0 = wmma_bf16(qf[c], kf0, s0);
        s1 = wmma_bf16(qf[c], kf1, s1);
      }
#pragma unroll
      for (int rr = 0; rr < 8; ++rr) {
        const int qr = q0 + rr + 8 * half;
        float a0 = s0[rr] * scale;
        float a1 = s1[rr] * scale;
        if (k0 + ln > qr)      a0 = -INFINITY;
        if (k0 + 16 + ln > qr) a1 = -INFINITY;
        float mv = fmaxf(a0, a1);
#pragma unroll
        for (int off = 1; off < 16; off <<= 1) mv = fmaxf(mv, __shfl_xor(mv, off, 32));
        const float mn = fmaxf(mrow[rr], mv);
        const float f  = expf(mrow[rr] - mn);
        const float p0 = expf(a0 - mn);
        const float p1 = expf(a1 - mn);
        float rs = p0 + p1;
#pragma unroll
        for (int off = 1; off < 16; off <<= 1) rs += __shfl_xor(rs, off, 32);
        lrow[rr] = lrow[rr] * f + rs;
        mrow[rr] = mn;
#pragma unroll
        for (int vt = 0; vt < 8; ++vt) acc[vt][rr] *= f;
        ps[wave][(rr + 8 * half) * 32 + ln]      = (__bf16)p0;
        ps[wave][(rr + 8 * half) * 32 + 16 + ln] = (__bf16)p1;
      }
      asm volatile("s_wait_dscnt 0" ::: "memory");  // CDNA5 split DS counter
      v16bf pf = load_frag_a(ps[wave], 32, lane);
#pragma unroll
      for (int vt = 0; vt < 8; ++vt) {
        v16bf vf = load_frag_b(vT + (vt * 16) * 32, 32, lane);
        acc[vt] = wmma_bf16(pf, vf, acc[vt]);
      }
    }
    __syncthreads();
  }

#pragma unroll
  for (int rr = 0; rr < 8; ++rr) {
    const float inv = 1.0f / lrow[rr];
    const int qr = q0 + rr + 8 * half;
#pragma unroll
    for (int vt = 0; vt < 8; ++vt)
      ao[(rowbase + qr) * (size_t)(NH_ * VD_) + h * VD_ + vt * 16 + ln] =
          (__bf16)(acc[vt][rr] * inv);
  }
}

// ---------------------------------------------------------------------------
// Residual + LayerNorm: y = LN(aop + rnn) * g + b  ->  bf16 (LM-head input)
// ---------------------------------------------------------------------------
__global__ void residual_ln_kernel(const float* __restrict__ aop,
                                   const float* __restrict__ rnn,
                                   const float* __restrict__ g,
                                   const float* __restrict__ bta,
                                   __bf16* __restrict__ y) {
  __shared__ float red[256];
  const int row = blockIdx.x, tid = threadIdx.x;
  const size_t base = (size_t)row * H_;

  float v[8];
  float s = 0.0f;
#pragma unroll
  for (int i = 0; i < 8; ++i) {
    const int c = tid + i * 256;
    const float x = aop[base + c] + rnn[base + c];
    v[i] = x;
    s += x;
  }
  red[tid] = s; __syncthreads();
  for (int off = 128; off > 0; off >>= 1) {
    if (tid < off) red[tid] += red[tid + off];
    __syncthreads();
  }
  const float mu = red[0] / (float)H_;
  __syncthreads();

  float sq = 0.0f;
#pragma unroll
  for (int i = 0; i < 8; ++i) { const float d = v[i] - mu; sq += d * d; }
  red[tid] = sq; __syncthreads();
  for (int off = 128; off > 0; off >>= 1) {
    if (tid < off) red[tid] += red[tid + off];
    __syncthreads();
  }
  const float rinv = rsqrtf(red[0] / (float)H_ + 1e-5f);

#pragma unroll
  for (int i = 0; i < 8; ++i) {
    const int c = tid + i * 256;
    y[base + c] = (__bf16)((v[i] - mu) * rinv * g[c] + bta[c]);
  }
}

// ---------------------------------------------------------------------------
// Host launcher
// ---------------------------------------------------------------------------
extern "C" void kernel_launch(void* const* d_in, const int* in_sizes, int n_in,
                              void* d_out, int out_size, void* d_ws, size_t ws_size,
                              hipStream_t stream) {
  (void)in_sizes; (void)n_in; (void)out_size; (void)ws_size;

  const int*   tokens = (const int*)  d_in[0];
  const float* E      = (const float*)d_in[1];
  const float* Wi     = (const float*)d_in[2];
  const float* bi     = (const float*)d_in[3];
  const float* Wh     = (const float*)d_in[4];
  const float* Wo     = (const float*)d_in[5];
  const float* bo     = (const float*)d_in[6];
  const float* Wqkv   = (const float*)d_in[7];
  const float* Wao    = (const float*)d_in[8];
  const float* bao    = (const float*)d_in[9];
  const float* ln_g   = (const float*)d_in[10];
  const float* ln_b   = (const float*)d_in[11];
  const float* Wlm    = (const float*)d_in[12];
  const float* blm    = (const float*)d_in[13];
  float* logits = (float*)d_out;

  // workspace carve-up (256B aligned regions)
  char* wp = (char*)d_ws;
  auto carve = [&](size_t bytes) -> void* {
    void* p = (void*)wp;
    wp += (bytes + 255) & ~(size_t)255;
    return p;
  };
  __bf16* xb     = (__bf16*)carve((size_t)M_ * DE_ * 2);
  __bf16* Wi_b   = (__bf16*)carve((size_t)H_ * DE_ * 2);
  __bf16* Wh_b   = (__bf16*)carve((size_t)H_ * H_ * 2);
  __bf16* Wo_b   = (__bf16*)carve((size_t)H_ * H_ * 2);
  __bf16* Wqkv_b = (__bf16*)carve((size_t)QKV_ * H_ * 2);
  __bf16* Wao_b  = (__bf16*)carve((size_t)H_ * H_ * 2);
  __bf16* Wlm_b  = (__bf16*)carve((size_t)V_ * H_ * 2);
  float*  xi     = (float*) carve((size_t)M_ * H_ * 4);
  __bf16* hbuf   = (__bf16*)carve((size_t)2 * 16 * H_ * 2);
  __bf16* hs     = (__bf16*)carve((size_t)M_ * H_ * 2);
  float*  rnn_f  = (float*) carve((size_t)M_ * H_ * 4);
  __bf16* rnn_b  = (__bf16*)carve((size_t)M_ * H_ * 2);
  __bf16* qkv_b  = (__bf16*)carve((size_t)M_ * QKV_ * 2);
  __bf16* ao_b   = (__bf16*)carve((size_t)M_ * (NH_ * VD_) * 2);
  float*  aop_f  = (float*) carve((size_t)M_ * H_ * 4);
  __bf16* y_b    = (__bf16*)carve((size_t)M_ * H_ * 2);
  unsigned int* bar = (unsigned int*)carve(256);

  // 1) weight conversion f32 -> bf16
  cvt_f32_bf16_kernel<<<1024, 256, 0, stream>>>(Wi,   Wi_b,   (size_t)H_ * DE_);
  cvt_f32_bf16_kernel<<<1024, 256, 0, stream>>>(Wh,   Wh_b,   (size_t)H_ * H_);
  cvt_f32_bf16_kernel<<<1024, 256, 0, stream>>>(Wo,   Wo_b,   (size_t)H_ * H_);
  cvt_f32_bf16_kernel<<<1024, 256, 0, stream>>>(Wqkv, Wqkv_b, (size_t)QKV_ * H_);
  cvt_f32_bf16_kernel<<<1024, 256, 0, stream>>>(Wao,  Wao_b,  (size_t)H_ * H_);
  cvt_f32_bf16_kernel<<<2048, 256, 0, stream>>>(Wlm,  Wlm_b,  (size_t)V_ * H_);

  // 2) embedding gather (f32 -> bf16 rows)
  embed_gather_kernel<<<M_, 256, 0, stream>>>(tokens, E, xb);

  // 3) xi = x @ Wi^T + bi  (f32 out, feeds RNN)
  gemm_bf16_kernel<true><<<dim3(H_ / 128, M_ / 128), 256, 0, stream>>>(
      xb, Wi_b, bi, xi, (__bf16*)nullptr, M_, H_, DE_);

  // 4) zero RNN state double-buffer + grid-barrier counter
  zero_u32_kernel<<<(2 * 16 * H_ * 2 / 4 + 255) / 256, 256, 0, stream>>>(
      (unsigned int*)hbuf, 2 * 16 * H_ * 2 / 4);
  zero_u32_kernel<<<1, 64, 0, stream>>>(bar, 64);

  // 5) sequential RNN scan (persistent, WMMA per step, grid barrier)
  rnn_scan_kernel<<<H_ / 16, 32, 0, stream>>>(xi, Wh_b, hbuf, hs, bar);

  // 6) rnn_out = hs @ Wo^T + bo (f32 for residual + bf16 for QKV input)
  gemm_bf16_kernel<true><<<dim3(H_ / 128, M_ / 128), 256, 0, stream>>>(
      hs, Wo_b, bo, rnn_f, rnn_b, M_, H_, H_);

  // 7) qkv = rnn_out @ Wqkv^T (bf16 out)
  gemm_bf16_kernel<false><<<dim3(QKV_ / 128, M_ / 128), 256, 0, stream>>>(
      rnn_b, Wqkv_b, (const float*)nullptr, (float*)nullptr, qkv_b, M_, QKV_, H_);

  // 8) causal flash attention -> ao (bf16)
  attn_kernel<<<dim3(B_ * NH_, T_ / 64), 128, 0, stream>>>(qkv_b, ao_b);

  // 9) ao_proj = ao @ Wao^T + bao (f32)
  gemm_bf16_kernel<true><<<dim3(H_ / 128, M_ / 128), 256, 0, stream>>>(
      ao_b, Wao_b, bao, aop_f, (__bf16*)nullptr, M_, H_, NH_ * VD_);

  // 10) y = LN(ao_proj + rnn_out) * g + b  -> bf16
  residual_ln_kernel<<<M_, 256, 0, stream>>>(aop_f, rnn_f, ln_g, ln_b, y_b);

  // 11) logits = y @ Wlm^T + blm (f32 into d_out)
  gemm_bf16_kernel<true><<<dim3(V_ / 128, M_ / 128), 256, 0, stream>>>(
      y_b, Wlm_b, blm, logits, (__bf16*)nullptr, M_, V_, H_);
}
